// EventDenoisingMamba_12300786336206
// MI455X (gfx1250) — compile-verified, matrix operand order checked by reading, not run on hardware
//
#include <hip/hip_runtime.h>
#include <hip/hip_bf16.h>

#define BSZ   8
#define LSEQ  4096
#define FEATN 11
#define DM    256
#define DI    512
#define DSN   16
#define DCN   4
#define DTRN  16
#define NLAY  4
#define ROWS  (BSZ * LSEQ)   // 32768

typedef __attribute__((ext_vector_type(16))) __bf16 v16bf;
typedef __attribute__((ext_vector_type(8)))  float  v8f;
typedef unsigned int u32x4 __attribute__((ext_vector_type(4)));
typedef int          i32x4 __attribute__((ext_vector_type(4)));
typedef int          i32x8 __attribute__((ext_vector_type(8)));

union Frag { v16bf v; uint4 u[2]; };

// ---------------------------------------------------------------- converts
__global__ void cvt_f32_bf16(const float* __restrict__ s, __bf16* __restrict__ d, int n) {
  int i = blockIdx.x * blockDim.x + threadIdx.x;
  if (i < n) d[i] = (__bf16)s[i];
}

// ---------------------------------------------------------------- embed (bf16 out)
__global__ __launch_bounds__(DM)
void embed_kernel(const float* __restrict__ feat, const float* __restrict__ w,
                  const float* __restrict__ b, __bf16* __restrict__ x) {
  const int row = blockIdx.x;
  const int dm  = threadIdx.x;
  const float* fr = feat + (size_t)row * FEATN;
  const float* wr = w + (size_t)dm * FEATN;
  float acc = b[dm];
#pragma unroll
  for (int f = 0; f < FEATN; ++f) acc += fr[f] * wr[f];
  x[(size_t)row * DM + dm] = (__bf16)acc;
}

// ---------------------------------------------------------------- WMMA GEMM
// C[M,N] = A[M,K](bf16 row-major) @ W[N,K](bf16 row-major)^T
// block = 256 threads (8 waves); wave computes 64 rows x (16*NT) cols.
// B tile (16*NT x K, <=32KB) staged in LDS once per block via TDM
// (tensor_load_to_lds + s_wait_tensorcnt) with a cooperative-copy fallback;
// fragments then come from ds_load_b128 while A streams via global b128.
template <int NT, bool OUTBF>
__global__ __launch_bounds__(256)
void wmma_gemm(const __bf16* __restrict__ A, const __bf16* __restrict__ W,
               void* __restrict__ Cout, int N, int K) {
  __shared__ uint4 Blds[2048];                 // 32 KB: up to 32 rows x 512 K bf16

  const int lane = threadIdx.x & 31;
  const int wave = threadIdx.x >> 5;
  const int half = lane >> 4;                  // K-group half select
  const int idx  = lane & 15;                  // row (A) / col (B)

  const int n0 = blockIdx.x * (16 * NT);
  const int m0 = blockIdx.y * 512 + wave * 64;

  const __bf16* wbase = W + (size_t)n0 * K;
  const int     ncols = 16 * NT;

#if __has_builtin(__builtin_amdgcn_tensor_load_to_lds)
  if (wave == 0) {
    // D# (CDNA5 ISA ch.8): contiguous 1-row tile of ncols*K bf16 elements.
    const unsigned lds_addr = (unsigned)(size_t)&Blds[0];
    const unsigned long long ga = (unsigned long long)(size_t)wbase;
    const unsigned td0 = (unsigned)(ncols * K);          // elements, data_size=2B
    u32x4 g0; i32x8 g1; i32x4 g2, g3;
    g0.x = 1u;                                           // count=1, user descriptor
    g0.y = lds_addr;                                     // lds_addr [63:32]
    g0.z = (unsigned)(ga & 0xFFFFFFFFu);                 // global_addr [95:64]
    g0.w = (unsigned)((ga >> 32) & 0x01FFFFFFu) | (2u << 30);  // addr hi + type=2
    g1[0] = (int)(1u << 16);                             // data_size=1 (2 bytes)
    g1[1] = (int)((td0 & 0xFFFFu) << 16);                // tensor_dim0 lo16 @ [63:48]
    g1[2] = (int)((td0 >> 16) | (1u << 16));             // tensor_dim0 hi16, tensor_dim1=1
    g1[3] = (int)((td0 & 0xFFFFu) << 16);                // tile_dim0 @ [127:112]
    g1[4] = (int)1u;                                     // tile_dim1=1, tile_dim2=0
    g1[5] = (int)td0;                                    // tensor_dim0_stride lo32
    g1[6] = 0; g1[7] = 0;
    g2 = (i32x4){0, 0, 0, 0};
    g3 = (i32x4){0, 0, 0, 0};
#if defined(__clang_major__) && __clang_major__ >= 23
    i32x8 g4 = (i32x8){0, 0, 0, 0, 0, 0, 0, 0};
    __builtin_amdgcn_tensor_load_to_lds(g0, g1, g2, g3, g4, 0);
#else
    __builtin_amdgcn_tensor_load_to_lds(g0, g1, g2, g3, 0);
#endif
    __builtin_amdgcn_s_wait_tensorcnt(0);
  }
  __syncthreads();
#else
  {
    const uint4* gw = (const uint4*)wbase;
    const int nu4 = (ncols * K) >> 3;
    for (int i = threadIdx.x; i < nu4; i += 256) Blds[i] = gw[i];
    __syncthreads();
  }
#endif

  const __bf16* ap[4];
#pragma unroll
  for (int t = 0; t < 4; ++t) ap[t] = A + (size_t)(m0 + 16 * t + idx) * K;
  const uint4* bb[NT];
#pragma unroll
  for (int j = 0; j < NT; ++j) bb[j] = Blds + (((16 * j + idx) * K) >> 3);

  v8f acc[4][NT];
#pragma unroll
  for (int t = 0; t < 4; ++t)
#pragma unroll
    for (int j = 0; j < NT; ++j) acc[t][j] = (v8f){};

  const int klo = 8 * half;          // first 16B run within the 32-wide K step
  const int khi = 16 + 8 * half;     // second 16B run

  for (int kk = 0; kk < K; kk += 32) {
    Frag bf[NT], af[4];
    const int bidx = (kk >> 3) + half;       // uint4 index of the klo run
#pragma unroll
    for (int j = 0; j < NT; ++j) {
      bf[j].u[0] = bb[j][bidx];              // ds_load_b128
      bf[j].u[1] = bb[j][bidx + 2];
    }
#pragma unroll
    for (int t = 0; t < 4; ++t) {
      af[t].u[0] = *(const uint4*)(ap[t] + kk + klo);   // global_load_b128
      af[t].u[1] = *(const uint4*)(ap[t] + kk + khi);
    }
#pragma unroll
    for (int t = 0; t < 4; ++t)
#pragma unroll
      for (int j = 0; j < NT; ++j)
        acc[t][j] = __builtin_amdgcn_wmma_f32_16x16x32_bf16(
            false, af[t].v, false, bf[j].v, (short)0, acc[t][j], false, false);
  }

#pragma unroll
  for (int t = 0; t < 4; ++t)
#pragma unroll
    for (int j = 0; j < NT; ++j) {
      const int col = n0 + 16 * j + idx;
#pragma unroll
      for (int v = 0; v < 8; ++v) {
        const size_t off = (size_t)(m0 + 16 * t + v + 8 * half) * N + col;
        if (OUTBF) ((__bf16*)Cout)[off] = (__bf16)acc[t][j][v];
        else       ((float*)Cout)[off]  = acc[t][j][v];
      }
    }
}

// ---------------------------------------------------------------- causal depthwise conv + silu
__global__ __launch_bounds__(256)
void conv_silu_kernel(const float* __restrict__ xz, const float* __restrict__ cw,
                      const float* __restrict__ cb, float* __restrict__ xc,
                      __bf16* __restrict__ xcb) {
  const int t   = blockIdx.x * blockDim.x + threadIdx.x;
  const int d   = t & (DI - 1);
  const int row = t >> 9;
  const int l   = row & (LSEQ - 1);
  const float* cwd = cw + (size_t)d * DCN;
  float acc = cb[d];
#pragma unroll
  for (int k = 0; k < DCN; ++k) {
    const int ls = l + k - (DCN - 1);
    if (ls >= 0) acc += xz[(size_t)(row + k - (DCN - 1)) * (2 * DI) + d] * cwd[k];
  }
  const float s = acc / (1.f + __expf(-acc));
  xc[(size_t)row * DI + d]  = s;
  xcb[(size_t)row * DI + d] = (__bf16)s;
}

// ---------------------------------------------------------------- fused selective scan
// lane = state s (16 lanes per (b,d)); dt-projection + softplus fused via shuffle
// reduction; epilogue (y + xc*D)*silu(z) written as bf16 for the out_proj GEMM.
__global__ __launch_bounds__(256)
void scan_kernel(const float* __restrict__ xc, __bf16* __restrict__ ybf,
                 const float* __restrict__ xz, const float* __restrict__ xdbl,
                 const float* __restrict__ dtp_w, const float* __restrict__ dtp_b,
                 const float* __restrict__ A_log, const float* __restrict__ Dp) {
  const int t  = blockIdx.x * blockDim.x + threadIdx.x;
  const int s  = t & (DSN - 1);
  const int bd = t >> 4;
  const int d  = bd & (DI - 1);
  const int b  = bd >> 9;

  const float a   = -__expf(A_log[(size_t)d * DSN + s]);
  const float wd  = dtp_w[(size_t)d * DTRN + s];
  const float dtb = dtp_b[d];
  const float Dd  = Dp[d];
  float h = 0.f;

  size_t row = (size_t)b * LSEQ;
  for (int l = 0; l < LSEQ; ++l, ++row) {
    const float* xrow = xdbl + row * (DTRN + 2 * DSN);
    float p = xrow[s] * wd;
#pragma unroll
    for (int m = 1; m < DSN; m <<= 1) p += __shfl_xor(p, m, 32);
    const float xin = p + dtb;
    const float dt  = (xin > 20.f) ? xin : __logf(1.f + __expf(xin));

    const float Bv  = xrow[DTRN + s];
    const float Cv  = xrow[DTRN + DSN + s];
    const float xcv = xc[row * DI + d];

    h = h * __expf(dt * a) + dt * Bv * xcv;

    float yp = h * Cv;
#pragma unroll
    for (int m = 1; m < DSN; m <<= 1) yp += __shfl_xor(yp, m, 32);

    if (s == 0) {
      const float z = xz[row * (2 * DI) + DI + d];
      const float g = z / (1.f + __expf(-z));
      ybf[row * DI + d] = (__bf16)((yp + xcv * Dd) * g);
    }
  }
}

// ---------------------------------------------------------------- head (bf16 in)
__global__ __launch_bounds__(256)
void head_kernel(const __bf16* __restrict__ x, const float* __restrict__ hw,
                 const float* __restrict__ hb, float* __restrict__ out) {
  const int lane = threadIdx.x & 31;
  const int wave = threadIdx.x >> 5;
  const int row  = blockIdx.x * 8 + wave;
  const __bf16* xr = x + (size_t)row * DM;
  float acc = 0.f;
#pragma unroll
  for (int i = 0; i < DM / 32; ++i) acc += (float)xr[lane + 32 * i] * hw[lane + 32 * i];
#pragma unroll
  for (int m = 1; m < 32; m <<= 1) acc += __shfl_xor(acc, m, 32);
  if (lane == 0) out[row] = 1.f / (1.f + __expf(-(acc + hb[0])));
}

// ---------------------------------------------------------------- launcher
extern "C" void kernel_launch(void* const* d_in, const int* in_sizes, int n_in,
                              void* d_out, int out_size, void* d_ws, size_t ws_size,
                              hipStream_t stream) {
  const float* features = (const float*)d_in[0];
  const float* emb_w    = (const float*)d_in[1];
  const float* emb_b    = (const float*)d_in[2];
  const float* in_w     = (const float*)d_in[3];
  const float* conv_w   = (const float*)d_in[4];
  const float* conv_b   = (const float*)d_in[5];
  const float* xp_w     = (const float*)d_in[6];
  const float* dtp_w    = (const float*)d_in[7];
  const float* dtp_b    = (const float*)d_in[8];
  const float* A_log    = (const float*)d_in[9];
  const float* Dp       = (const float*)d_in[10];
  const float* out_w    = (const float*)d_in[11];
  const float* head_w   = (const float*)d_in[12];
  const float* head_b   = (const float*)d_in[13];

  char* ws = (char*)d_ws;
  __bf16* xbf   = (__bf16*)(ws + 0);             // 32768*256*2  = 16.8 MB
  float*  xzbuf = (float*)(ws + 16777216ull);    // 32768*1024*4 = 134 MB
  float*  xcbuf = (float*)(ws + 150994944ull);   // 32768*512*4  = 67 MB
  __bf16* xcbf  = (__bf16*)(ws + 218103808ull);  // 32768*512*2  = 33.5 MB (reused as y_bf)
  float*  xdbuf = (float*)(ws + 251658240ull);   // 32768*48*4   = 6.3 MB
  __bf16* w_in  = (__bf16*)(ws + 257949696ull);  // 4*1024*256*2
  __bf16* w_xp  = (__bf16*)(ws + 260046848ull);  // 4*48*512*2
  __bf16* w_out = (__bf16*)(ws + 260243456ull);  // 4*256*512*2

  const int n_in_w  = NLAY * 2 * DI * DM;
  const int n_xp_w  = NLAY * 48 * DI;
  const int n_out_w = NLAY * DM * DI;
  cvt_f32_bf16<<<(n_in_w  + 255) / 256, 256, 0, stream>>>(in_w,  w_in,  n_in_w);
  cvt_f32_bf16<<<(n_xp_w  + 255) / 256, 256, 0, stream>>>(xp_w,  w_xp,  n_xp_w);
  cvt_f32_bf16<<<(n_out_w + 255) / 256, 256, 0, stream>>>(out_w, w_out, n_out_w);

  embed_kernel<<<ROWS, DM, 0, stream>>>(features, emb_w, emb_b, xbf);

  for (int i = 0; i < NLAY; ++i) {
    // in_proj: xz = x @ in_w^T   (M=32768, N=1024, K=256), f32 out
    wmma_gemm<2, false><<<dim3(1024 / 32, ROWS / 512), 256, 0, stream>>>(
        xbf, w_in + (size_t)i * 1024 * 256, xzbuf, 1024, 256);

    // causal conv + silu -> xc (f32) + xc (bf16)
    conv_silu_kernel<<<(ROWS * DI) / 256, 256, 0, stream>>>(
        xzbuf, conv_w + (size_t)i * DI * DCN, conv_b + (size_t)i * DI, xcbuf, xcbf);

    // x_proj: xdbl = xc @ xp_w^T (M=32768, N=48, K=512), f32 out
    wmma_gemm<1, false><<<dim3(48 / 16, ROWS / 512), 256, 0, stream>>>(
        xcbf, w_xp + (size_t)i * 48 * DI, xdbuf, 48, 512);

    // fused dt-proj + selective scan + gating -> y (bf16, overwrites xcbf)
    scan_kernel<<<(BSZ * DI * DSN) / 256, 256, 0, stream>>>(
        xcbuf, xcbf, xzbuf, xdbuf,
        dtp_w + (size_t)i * DI * DTRN, dtp_b + (size_t)i * DI,
        A_log + (size_t)i * DI * DSN,  Dp + (size_t)i * DI);

    // out_proj: x = y @ out_w^T  (M=32768, N=256, K=512), bf16 out
    wmma_gemm<2, true><<<dim3(256 / 32, ROWS / 512), 256, 0, stream>>>(
        xcbf, w_out + (size_t)i * DM * DI, xbf, 256, 512);
  }

  head_kernel<<<ROWS / 8, 256, 0, stream>>>(xbf, head_w, head_b, (float*)d_out);
}
